// DecoderBlock_30837865185850
// MI455X (gfx1250) — compile-verified, matrix operand order checked by reading, not run on hardware
//
#include <hip/hip_runtime.h>
#include <stdint.h>

// ---------------------------------------------------------------------------
// Decoder block for MI455X (gfx1250): all GEMM-shaped work through
// v_wmma_f32_16x16x32_bf16 (wave32). GEMM weights pre-tiled to B-fragment
// order and staged via global_load_async_to_lds_b128 (ASYNCcnt) with double
// buffering; flash-attention consumes V pre-transposed by the V-proj GEMM.
// ---------------------------------------------------------------------------

typedef __bf16 v16bf __attribute__((ext_vector_type(16)));
typedef float  v8f   __attribute__((ext_vector_type(8)));
typedef unsigned short u16;

union Frag {
  v16bf v;
  uint4 q[2];
  u16   u[16];
};

__device__ __forceinline__ u16 f2bf(float f) {
  unsigned int u = __float_as_uint(f);
  u += 0x7FFFu + ((u >> 16) & 1u);   // round-to-nearest-even
  return (u16)(u >> 16);
}

__device__ __forceinline__ v8f wmma_bf16(const Frag& a, const Frag& b, v8f c) {
  return __builtin_amdgcn_wmma_f32_16x16x32_bf16(false, a.v, false, b.v,
                                                 (short)0, c, false, false);
}

// Async global->LDS copy of 16B per lane; tracked by ASYNCcnt.
// `lds` is passed (unused in the template) so the LDS array escapes into the
// asm: the "memory" clobber then legally covers it and the compiler cannot
// fold the subsequent ds_load_b128 reads to undef.
__device__ __forceinline__ void async_cp16(unsigned lds_off, const void* g,
                                           void* lds) {
  asm volatile("global_load_async_to_lds_b128 %0, %1, off"
               :: "v"(lds_off), "v"((unsigned long long)(uintptr_t)g),
                  "v"(lds)
               : "memory");
}
__device__ __forceinline__ void wait_async0(void* lds) {
  asm volatile("s_wait_asynccnt 0" :: "v"(lds) : "memory");
}

// tiled weight index: [n>>6][k>>5] tiles of [n&63][k&31]
__device__ __forceinline__ size_t tile_idx(int k, int n, int K) {
  return ((size_t)(n >> 6) * (K >> 5) + (k >> 5)) * 2048 +
         (size_t)(n & 63) * 32 + (k & 31);
}

// ------------------------------- prep kernels ------------------------------

__global__ void k_f32_to_bf16(const float* __restrict__ in,
                              u16* __restrict__ out, long long n) {
  long long i = (long long)blockIdx.x * blockDim.x + threadIdx.x;
  long long stride = (long long)gridDim.x * blockDim.x;
  for (; i < n; i += stride) out[i] = f2bf(in[i]);
}

// plain [K][N] f32 -> bf16 tiled
__global__ void k_pack_w_plain(const float* __restrict__ in,
                               u16* __restrict__ out, int K, int N) {
  long long n = (long long)K * N;
  long long i = (long long)blockIdx.x * blockDim.x + threadIdx.x;
  long long stride = (long long)gridDim.x * blockDim.x;
  for (; i < n; i += stride) {
    int k = (int)(i / N), c = (int)(i % N);
    out[tile_idx(k, c, K)] = f2bf(in[i]);
  }
}

// per-head [H=16][C=1024][HS=64] f32 -> bf16 tiled, n = h*64+d, k = c
__global__ void k_pack_w_head(const float* __restrict__ in,
                              u16* __restrict__ out) {
  long long n = 16LL * 1024 * 64;
  long long i = (long long)blockIdx.x * blockDim.x + threadIdx.x;
  long long stride = (long long)gridDim.x * blockDim.x;
  for (; i < n; i += stride) {
    int d = (int)(i & 63);
    int c = (int)((i >> 6) & 1023);
    int h = (int)(i >> 16);
    out[tile_idx(c, h * 64 + d, 1024)] = f2bf(in[i]);
  }
}

// block per row, C=1024, 256 threads x 4 elems; bf16 output
__global__ void k_layernorm_bf16(const float* __restrict__ x,
                                 const float* __restrict__ g,
                                 const float* __restrict__ bta,
                                 u16* __restrict__ out) {
  __shared__ float s_sum[256];
  __shared__ float s_sq[256];
  int row = blockIdx.x, tid = threadIdx.x;
  const float* xr = x + (size_t)row * 1024;
  float4 v = *(const float4*)(xr + tid * 4);
  float s = v.x + v.y + v.z + v.w;
  float q = v.x * v.x + v.y * v.y + v.z * v.z + v.w * v.w;
  s_sum[tid] = s; s_sq[tid] = q;
  __syncthreads();
  for (int off = 128; off > 0; off >>= 1) {
    if (tid < off) { s_sum[tid] += s_sum[tid + off]; s_sq[tid] += s_sq[tid + off]; }
    __syncthreads();
  }
  float mu = s_sum[0] * (1.0f / 1024.0f);
  float var = s_sq[0] * (1.0f / 1024.0f) - mu * mu;
  float rstd = rsqrtf(var + 1e-5f);
  float4 gg = *(const float4*)(g + tid * 4);
  float4 bb = *(const float4*)(bta + tid * 4);
  float y0 = (v.x - mu) * rstd * gg.x + bb.x;
  float y1 = (v.y - mu) * rstd * gg.y + bb.y;
  float y2 = (v.z - mu) * rstd * gg.z + bb.z;
  float y3 = (v.w - mu) * rstd * gg.w + bb.w;
  uint2 o;
  o.x = (unsigned)f2bf(y0) | ((unsigned)f2bf(y1) << 16);
  o.y = (unsigned)f2bf(y2) | ((unsigned)f2bf(y3) << 16);
  *(uint2*)(out + (size_t)row * 1024 + tid * 4) = o;
}

// ------------------------------- WMMA GEMM ---------------------------------
// O[M,N] = act(A_bf16[M,K] @ W[K,N] + bias); W pre-tiled. 8 waves/block,
// block tile 128(M) x 64(N); K staged 64-deep, double-buffered async LDS.
// out_mode: 0 = f32 (+opt resid), 1 = bf16 row-major, 2 = bf16 [b,h,d,s].
__global__ __launch_bounds__(256) void k_gemm_wmma(
    const u16* __restrict__ A, const u16* __restrict__ Wt,
    const float* __restrict__ bias, const float* __restrict__ resid,
    float* __restrict__ outf, u16* __restrict__ outb,
    int M, int N, int K, int relu, int out_mode) {
  // single shared var => LDS offset 0 for async stores
  __shared__ __align__(16) u16 wlds[2][2][64 * 32];  // [buf][ktile][n*32+k]
  void* ldsp = (void*)&wlds[0][0][0];
  int tid = threadIdx.x, lane = tid & 31, wave = tid >> 5;
  int n0 = blockIdx.x * 64;
  int m0 = blockIdx.y * 128 + wave * 16;
  int ml = lane & 15;
  int hi = (lane >> 4) * 8;  // half-wave K/M offset per WMMA bf16 layout

  const u16* wpanel = Wt + (size_t)(n0 >> 6) * (K >> 5) * 2048;
  const u16* ar = A + (size_t)(m0 + ml) * K;

  v8f acc[4];
#pragma unroll
  for (int t = 0; t < 4; ++t)
#pragma unroll
    for (int j = 0; j < 8; ++j) acc[t][j] = 0.0f;

  int nst = K >> 6;  // stages of 64 K
  // prologue: stage 0 into buf 0 (8KB = 256 threads x 2 x 16B)
  {
    const u16* src = wpanel;  // k-tiles 0,1 contiguous
    async_cp16((unsigned)(tid * 16), src + tid * 8, ldsp);
    async_cp16((unsigned)(tid * 16 + 4096), src + tid * 8 + 2048, ldsp);
  }

  for (int st = 0; st < nst; ++st) {
    int buf = st & 1;
    wait_async0(ldsp);
    __syncthreads();
    if (st + 1 < nst) {
      const u16* src = wpanel + (size_t)(st + 1) * 4096;
      unsigned base = (unsigned)((1 - buf) * 8192);
      async_cp16(base + tid * 16, src + tid * 8, ldsp);
      async_cp16(base + tid * 16 + 4096, src + tid * 8 + 2048, ldsp);
    }
    __builtin_prefetch(ar + (st + 1) * 64, 0, 3);
#pragma unroll
    for (int kk = 0; kk < 2; ++kk) {
      Frag a;
      const u16* arc = ar + st * 64 + kk * 32;
      a.q[0] = *(const uint4*)(arc + hi);
      a.q[1] = *(const uint4*)(arc + 16 + hi);
#pragma unroll
      for (int t = 0; t < 4; ++t) {
        Frag b;
        const u16* bp = &wlds[buf][kk][(t * 16 + ml) * 32 + hi];
        b.q[0] = *(const uint4*)bp;
        b.q[1] = *(const uint4*)(bp + 16);
        acc[t] = wmma_bf16(a, b, acc[t]);
      }
    }
    __syncthreads();
  }

#pragma unroll
  for (int t = 0; t < 4; ++t) {
    int col = n0 + t * 16 + ml;
    float bv = bias ? bias[col] : 0.0f;
#pragma unroll
    for (int j = 0; j < 8; ++j) {
      int row = m0 + j + hi;
      float v = acc[t][j] + bv;
      if (relu) v = fmaxf(v, 0.0f);
      if (out_mode == 0) {
        if (resid) v += resid[(size_t)row * N + col];
        outf[(size_t)row * N + col] = v;
      } else if (out_mode == 1) {
        outb[(size_t)row * N + col] = f2bf(v);
      } else {  // [b,h,d,s] for attention V (rows = b*2048+s, col = h*64+d)
        int bb2 = row >> 11, sr = row & 2047, hh = col >> 6, dd = col & 63;
        outb[(((size_t)bb2 * 16 + hh) * 64 + dd) * 2048 + sr] = f2bf(v);
      }
    }
  }
}

// ---------------------------- flash attention ------------------------------
// Q,K bf16 [B*rows, H*64]; VT bf16 [B,H,64,S]. Per wave: 16 q-rows of one
// (b,h); S streamed in 32-chunks; online softmax; P transposed via LDS.
__global__ __launch_bounds__(128) void k_attn_wmma(
    const u16* __restrict__ Q, const u16* __restrict__ Km,
    const u16* __restrict__ VT, u16* __restrict__ O,
    int T, int S, int causal) {
  __shared__ __align__(16) u16 plds[4][16 * 32];  // [m][s] per wave
  int tid = threadIdx.x, lane = tid & 31, wave = tid >> 5;
  int b = blockIdx.z, h = blockIdx.y;
  int t0 = blockIdx.x * 64 + wave * 16;
  int ml = lane & 15;
  int hi = (lane >> 4) * 8;
  u16* pl = plds[wave];
  const u16* vtb = VT + ((size_t)(b * 16 + h) * 64) * 2048;

  Frag qa[2];
  {
    const u16* qr = Q + (size_t)(b * T + t0 + ml) * 1024 + h * 64;
#pragma unroll
    for (int c = 0; c < 2; ++c) {
      qa[c].q[0] = *(const uint4*)(qr + c * 32 + hi);
      qa[c].q[1] = *(const uint4*)(qr + c * 32 + 16 + hi);
    }
  }

  v8f o[4];
  float mrow[8], lrow[8], alpha[8];
#pragma unroll
  for (int t = 0; t < 4; ++t)
#pragma unroll
    for (int j = 0; j < 8; ++j) o[t][j] = 0.0f;
#pragma unroll
  for (int j = 0; j < 8; ++j) { mrow[j] = -3.0e38f; lrow[j] = 0.0f; }

  int send = causal ? (t0 + 16) : S;
  int nch = (send + 31) >> 5;

  for (int ch = 0; ch < nch; ++ch) {
    int s0 = ch * 32;
    v8f sc[2];
#pragma unroll
    for (int t2 = 0; t2 < 2; ++t2) {
      v8f s;
#pragma unroll
      for (int j = 0; j < 8; ++j) s[j] = 0.0f;
      const u16* kr = Km + (size_t)(b * S + s0 + t2 * 16 + ml) * 1024 + h * 64;
#pragma unroll
      for (int c = 0; c < 2; ++c) {
        Frag kb;
        kb.q[0] = *(const uint4*)(kr + c * 32 + hi);
        kb.q[1] = *(const uint4*)(kr + c * 32 + 16 + hi);
        s = wmma_bf16(qa[c], kb, s);
      }
      int scol = s0 + t2 * 16 + ml;
#pragma unroll
      for (int j = 0; j < 8; ++j) {
        float v = s[j] * 0.125f;  // HS^-0.5
        if (causal && scol > (t0 + j + hi)) v = -1.0e30f;
        s[j] = v;
      }
      sc[t2] = s;
    }

#pragma unroll
    for (int j = 0; j < 8; ++j) {
      float t = fmaxf(sc[0][j], sc[1][j]);
#pragma unroll
      for (int d2 = 8; d2 > 0; d2 >>= 1) t = fmaxf(t, __shfl_xor(t, d2, 32));
      float mn = fmaxf(mrow[j], t);
      float a = __expf(mrow[j] - mn);
      float p0 = __expf(sc[0][j] - mn);
      float p1 = __expf(sc[1][j] - mn);
      sc[0][j] = p0; sc[1][j] = p1;
      float rs = p0 + p1;
#pragma unroll
      for (int d2 = 8; d2 > 0; d2 >>= 1) rs += __shfl_xor(rs, d2, 32);
      lrow[j] = lrow[j] * a + rs;
      mrow[j] = mn;
      alpha[j] = a;
    }
#pragma unroll
    for (int t = 0; t < 4; ++t)
#pragma unroll
      for (int j = 0; j < 8; ++j) o[t][j] *= alpha[j];

    // transpose P (C layout -> A layout) through LDS
#pragma unroll
    for (int t2 = 0; t2 < 2; ++t2)
#pragma unroll
      for (int j = 0; j < 8; ++j)
        pl[(j + hi) * 32 + t2 * 16 + ml] = f2bf(sc[t2][j]);
    asm volatile("s_wait_dscnt 0" ::: "memory");

    Frag pa;
    pa.q[0] = *(const uint4*)&pl[ml * 32 + hi];
    pa.q[1] = *(const uint4*)&pl[ml * 32 + 16 + hi];
#pragma unroll
    for (int t = 0; t < 4; ++t) {
      Frag vb;
      const u16* vr = vtb + (size_t)(t * 16 + ml) * 2048 + s0;
      vb.q[0] = *(const uint4*)(vr + hi);
      vb.q[1] = *(const uint4*)(vr + 16 + hi);
      o[t] = wmma_bf16(pa, vb, o[t]);
    }
  }

#pragma unroll
  for (int t = 0; t < 4; ++t)
#pragma unroll
    for (int j = 0; j < 8; ++j) {
      float v = o[t][j] / lrow[j];
      O[(size_t)(b * T + t0 + j + hi) * 1024 + h * 64 + t * 16 + ml] = f2bf(v);
    }
}

// ------------------------------- launcher ----------------------------------

extern "C" void kernel_launch(void* const* d_in, const int* in_sizes, int n_in,
                              void* d_out, int out_size, void* d_ws,
                              size_t ws_size, hipStream_t stream) {
  (void)in_sizes; (void)n_in; (void)out_size; (void)ws_size;
  const float* x_e   = (const float*)d_in[0];
  const float* x     = (const float*)d_in[1];
  const float* mq_w  = (const float*)d_in[2];
  const float* mq_b  = (const float*)d_in[3];
  const float* mk_w  = (const float*)d_in[4];
  const float* mk_b  = (const float*)d_in[5];
  const float* mv_w  = (const float*)d_in[6];
  const float* mv_b  = (const float*)d_in[7];
  const float* m_proj_w = (const float*)d_in[8];
  const float* m_proj_b = (const float*)d_in[9];
  const float* cq_w  = (const float*)d_in[10];
  const float* cq_b  = (const float*)d_in[11];
  const float* ck_w  = (const float*)d_in[12];
  const float* ck_b  = (const float*)d_in[13];
  const float* cv_w  = (const float*)d_in[14];
  const float* cv_b  = (const float*)d_in[15];
  const float* c_proj_w = (const float*)d_in[16];
  const float* c_proj_b = (const float*)d_in[17];
  const float* f_w1  = (const float*)d_in[18];
  const float* f_b1  = (const float*)d_in[19];
  const float* f_w2  = (const float*)d_in[20];
  const float* f_b2  = (const float*)d_in[21];
  const float* ln1_g = (const float*)d_in[22];
  const float* ln1_b = (const float*)d_in[23];
  const float* ln2_g = (const float*)d_in[24];
  const float* ln2_b = (const float*)d_in[25];
  const float* ln3_g = (const float*)d_in[26];
  const float* ln3_b = (const float*)d_in[27];

  constexpr size_t MB = 1024ull * 1024ull;
  char* ws = (char*)d_ws;
  u16* WQ  = (u16*)(ws + 0 * MB);    // 6 tiled head weights, 2MB each
  u16* WK  = (u16*)(ws + 2 * MB);
  u16* WV  = (u16*)(ws + 4 * MB);
  u16* WCQ = (u16*)(ws + 6 * MB);
  u16* WCK = (u16*)(ws + 8 * MB);
  u16* WCV = (u16*)(ws + 10 * MB);
  u16* MP  = (u16*)(ws + 12 * MB);   // m_proj tiled bf16
  u16* CP  = (u16*)(ws + 14 * MB);   // c_proj tiled bf16
  u16* FW1 = (u16*)(ws + 16 * MB);   // tiled, 8MB
  u16* FW2 = (u16*)(ws + 24 * MB);   // tiled, 8MB
  u16* XE  = (u16*)(ws + 32 * MB);   // x_e bf16 row-major, 8MB
  u16* ALN = (u16*)(ws + 40 * MB);   // LN output bf16, 8MB
  u16* QB  = (u16*)(ws + 48 * MB);
  u16* KB  = (u16*)(ws + 56 * MB);
  u16* VTB = (u16*)(ws + 64 * MB);   // V transposed [b,h,d,s]
  u16* HB  = (u16*)(ws + 72 * MB);   // attention output bf16
  float* X1 = (float*)(ws + 80 * MB);  // residual after self-attn, 16MB
  u16* F1  = (u16*)(ws + 96 * MB);   // FFN hidden bf16, 32MB
  float* XOUT = (float*)d_out;

  // --- weight / input prep (bf16, tiled for GEMM B-side) ---
  k_pack_w_head<<<1024, 256, 0, stream>>>(mq_w, WQ);
  k_pack_w_head<<<1024, 256, 0, stream>>>(mk_w, WK);
  k_pack_w_head<<<1024, 256, 0, stream>>>(mv_w, WV);
  k_pack_w_head<<<1024, 256, 0, stream>>>(cq_w, WCQ);
  k_pack_w_head<<<1024, 256, 0, stream>>>(ck_w, WCK);
  k_pack_w_head<<<1024, 256, 0, stream>>>(cv_w, WCV);
  k_pack_w_plain<<<1024, 256, 0, stream>>>(m_proj_w, MP, 1024, 1024);
  k_pack_w_plain<<<1024, 256, 0, stream>>>(c_proj_w, CP, 1024, 1024);
  k_pack_w_plain<<<2048, 256, 0, stream>>>(f_w1, FW1, 1024, 4096);
  k_pack_w_plain<<<2048, 256, 0, stream>>>(f_w2, FW2, 4096, 1024);
  k_f32_to_bf16<<<2048, 256, 0, stream>>>(x_e, XE, 4096LL * 1024);

  dim3 g1(1024 / 64, 4096 / 128);  // N=1024 GEMMs
  dim3 g2(4096 / 64, 4096 / 128);  // N=4096 GEMM (FFN up)
  dim3 ga(2048 / 64, 16, 2);       // attention

  // --- masked self-attention ---
  k_layernorm_bf16<<<4096, 256, 0, stream>>>(x, ln1_g, ln1_b, ALN);
  k_gemm_wmma<<<g1, 256, 0, stream>>>(ALN, WQ, mq_b, nullptr, nullptr, QB, 4096, 1024, 1024, 0, 1);
  k_gemm_wmma<<<g1, 256, 0, stream>>>(ALN, WK, mk_b, nullptr, nullptr, KB, 4096, 1024, 1024, 0, 1);
  k_gemm_wmma<<<g1, 256, 0, stream>>>(ALN, WV, mv_b, nullptr, nullptr, VTB, 4096, 1024, 1024, 0, 2);
  k_attn_wmma<<<ga, 128, 0, stream>>>(QB, KB, VTB, HB, 2048, 2048, 1);
  k_gemm_wmma<<<g1, 256, 0, stream>>>(HB, MP, m_proj_b, x, X1, nullptr, 4096, 1024, 1024, 0, 0);

  // --- cross-attention (KV from raw x_e) ---
  k_layernorm_bf16<<<4096, 256, 0, stream>>>(X1, ln2_g, ln2_b, ALN);
  k_gemm_wmma<<<g1, 256, 0, stream>>>(ALN, WCQ, cq_b, nullptr, nullptr, QB, 4096, 1024, 1024, 0, 1);
  k_gemm_wmma<<<g1, 256, 0, stream>>>(XE, WCK, ck_b, nullptr, nullptr, KB, 4096, 1024, 1024, 0, 1);
  k_gemm_wmma<<<g1, 256, 0, stream>>>(XE, WCV, cv_b, nullptr, nullptr, VTB, 4096, 1024, 1024, 0, 2);
  k_attn_wmma<<<ga, 128, 0, stream>>>(QB, KB, VTB, HB, 2048, 2048, 0);
  k_gemm_wmma<<<g1, 256, 0, stream>>>(HB, CP, c_proj_b, X1, XOUT, nullptr, 4096, 1024, 1024, 0, 0);

  // --- FFN ---
  k_layernorm_bf16<<<4096, 256, 0, stream>>>(XOUT, ln3_g, ln3_b, ALN);
  k_gemm_wmma<<<g2, 256, 0, stream>>>(ALN, FW1, f_b1, nullptr, nullptr, F1, 4096, 4096, 1024, 1, 1);
  k_gemm_wmma<<<g1, 256, 0, stream>>>(F1, FW2, f_b2, XOUT, XOUT, nullptr, 4096, 1024, 4096, 0, 0);
}
